// ImageRNN_78005196030244
// MI455X (gfx1250) — matrix operand order repaired and checked by reading, not compiled, not run on hardware
//
#include <hip/hip_runtime.h>

#define B_ 256
#define T_ 224
#define I_ 224
#define H_ 1024
#define O_ 4

// CDNA5 WMMA vector types (wave32)
typedef __attribute__((ext_vector_type(16))) __bf16 v16bf;
typedef __attribute__((ext_vector_type(8)))  float  v8f;
typedef __attribute__((ext_vector_type(4)))  unsigned int u32x4;

union FragU { u32x4 q[2]; v16bf v; };

// bf16 helpers (memory type = unsigned short, RNE convert)
__device__ __forceinline__ unsigned short f2bf(float f) {
    unsigned int u = __builtin_bit_cast(unsigned int, f);
    u += 0x7FFFu + ((u >> 16) & 1u);
    return (unsigned short)(u >> 16);
}
__device__ __forceinline__ float bf2f(unsigned short h) {
    unsigned int u = ((unsigned int)h) << 16;
    return __builtin_bit_cast(float, u);
}

// A-matrix 16x32 bf16 fragment (ISA 7.12.2): lane row = lane%16,
// half = lane/16: elements [0..7] = K k+half*8.. , [8..15] = K k+16+half*8..
__device__ __forceinline__ v16bf load_a_frag(const unsigned short* rowbase, int k, int half) {
    FragU u;
    const unsigned short* p = rowbase + k + half * 8;
    u.q[0] = *(const u32x4*)(p);
    u.q[1] = *(const u32x4*)(p + 16);
    return u.v;
}

// B-matrix 32x16 bf16 fragment: lane col = lane%16, half = lane/16:
// elements [0..7] = K k+half*8.. , [8..15] = K k+16+half*8.. when the
// weight is stored (N,K) row-major (i.e. W itself, since we compute x @ W^T).
__device__ __forceinline__ v16bf load_b_frag(const unsigned short* colbase, int k, int half) {
    FragU u;
    const unsigned short* p = colbase + k + half * 8;
    u.q[0] = *(const u32x4*)(p);
    u.q[1] = *(const u32x4*)(p + 16);
    return u.v;
}

__device__ __forceinline__ void gemm_phase(
    const unsigned short* __restrict__ a0, const unsigned short* __restrict__ a1,
    const unsigned short* __restrict__ b0, const unsigned short* __restrict__ b1,
    int K, int half, v8f& acc00, v8f& acc01, v8f& acc10, v8f& acc11)
{
    for (int k = 0; k < K; k += 32) {
        v16bf A0 = load_a_frag(a0, k, half);
        v16bf A1 = load_a_frag(a1, k, half);
        v16bf B0 = load_b_frag(b0, k, half);
        v16bf B1 = load_b_frag(b1, k, half);
        acc00 = __builtin_amdgcn_wmma_f32_16x16x32_bf16(false, A0, false, B0, (short)0, acc00, false, false);
        acc01 = __builtin_amdgcn_wmma_f32_16x16x32_bf16(false, A0, false, B1, (short)0, acc01, false, false);
        acc10 = __builtin_amdgcn_wmma_f32_16x16x32_bf16(false, A1, false, B0, (short)0, acc10, false, false);
        acc11 = __builtin_amdgcn_wmma_f32_16x16x32_bf16(false, A1, false, B1, (short)0, acc11, false, false);
    }
}

// One recurrence step, fused with the input projection:
//   h_out = tanh( x[:,t,:] @ W_ih^T + h_in @ W_hh^T + (b_ih+b_hh) )
// Wave computes a 32x32 output tile. Grid: 32 WGs x 8 waves covers 256x1024.
__global__ void __launch_bounds__(256) rnn_step_kernel(
    const unsigned short* __restrict__ xbf,   // (B, T, I) bf16
    const unsigned short* __restrict__ Wih,   // (H, I) bf16  (= B^T layout)
    const unsigned short* __restrict__ Whh,   // (H, H) bf16  (= B^T layout)
    const float*          __restrict__ bias,  // (H) b_ih+b_hh
    const unsigned short* __restrict__ h_in,  // (B, H) bf16
    unsigned short*       __restrict__ h_out, // (B, H) bf16
    int t)
{
    const int lane = threadIdx.x & 31;
    const int wave = threadIdx.x >> 5;
    const int lr   = lane & 15;
    const int half = lane >> 4;

    const int br = blockIdx.x >> 2;          // 0..7  (row block of 32)
    const int bc = blockIdx.x & 3;           // 0..3  (col block of 256)
    const int row0 = br * 32;
    const int n0   = bc * 256 + wave * 32;

    v8f acc00 = {}, acc01 = {}, acc10 = {}, acc11 = {};

    // Phase 1: x_t @ W_ih^T  (K = I_ = 224)
    {
        const unsigned short* a0 = xbf + (size_t)(row0 + lr)      * (T_ * I_) + (size_t)t * I_;
        const unsigned short* a1 = xbf + (size_t)(row0 + 16 + lr) * (T_ * I_) + (size_t)t * I_;
        const unsigned short* b0 = Wih + (size_t)(n0 + lr)      * I_;
        const unsigned short* b1 = Wih + (size_t)(n0 + 16 + lr) * I_;
        gemm_phase(a0, a1, b0, b1, I_, half, acc00, acc01, acc10, acc11);
    }

    // Phase 2: h @ W_hh^T  (K = H_ = 1024)
    {
        const unsigned short* a0 = h_in + (size_t)(row0 + lr)      * H_;
        const unsigned short* a1 = h_in + (size_t)(row0 + 16 + lr) * H_;
        const unsigned short* b0 = Whh + (size_t)(n0 + lr)      * H_;
        const unsigned short* b1 = Whh + (size_t)(n0 + 16 + lr) * H_;
        gemm_phase(a0, a1, b0, b1, H_, half, acc00, acc01, acc10, acc11);
    }

    // Epilogue: bias + tanh, store bf16. C/D layout: VGPR r -> row r + 8*half,
    // col = lane%16 (ISA 7.12.2).
    const float bias0 = bias[n0 + lr];
    const float bias1 = bias[n0 + 16 + lr];
    #pragma unroll
    for (int r = 0; r < 8; ++r) {
        const int rr0 = row0 + r + 8 * half;
        const int rr1 = rr0 + 16;
        h_out[(size_t)rr0 * H_ + n0 + lr]      = f2bf(tanhf(acc00[r] + bias0));
        h_out[(size_t)rr0 * H_ + n0 + 16 + lr] = f2bf(tanhf(acc01[r] + bias1));
        h_out[(size_t)rr1 * H_ + n0 + lr]      = f2bf(tanhf(acc10[r] + bias0));
        h_out[(size_t)rr1 * H_ + n0 + 16 + lr] = f2bf(tanhf(acc11[r] + bias1));
    }
}

__global__ void f32_to_bf16_kernel(const float* __restrict__ in,
                                   unsigned short* __restrict__ out, int n) {
    int i = blockIdx.x * blockDim.x + threadIdx.x;
    if (i < n) out[i] = f2bf(in[i]);
}

__global__ void zero32_kernel(unsigned int* __restrict__ p, int n) {
    int i = blockIdx.x * blockDim.x + threadIdx.x;
    if (i < n) p[i] = 0u;
}

__global__ void bias_combine_kernel(const float* __restrict__ a,
                                    const float* __restrict__ b,
                                    float* __restrict__ out, int n) {
    int i = blockIdx.x * blockDim.x + threadIdx.x;
    if (i < n) out[i] = a[i] + b[i];
}

// out[b,o] = sum_k h[b,k] * W_fc[o,k] + b_fc[o]   (B*O = 1024 dots of len 1024)
__global__ void fc_kernel(const unsigned short* __restrict__ h,
                          const float* __restrict__ Wfc,
                          const float* __restrict__ bfc,
                          float* __restrict__ out) {
    int idx = blockIdx.x * blockDim.x + threadIdx.x;
    if (idx >= B_ * O_) return;
    int b = idx >> 2;
    int o = idx & 3;
    const unsigned short* hr = h + (size_t)b * H_;
    const float* wr = Wfc + (size_t)o * H_;
    float s = bfc[o];
    for (int k = 0; k < H_; ++k) s += bf2f(hr[k]) * wr[k];
    out[idx] = s;
}

extern "C" void kernel_launch(void* const* d_in, const int* in_sizes, int n_in,
                              void* d_out, int out_size, void* d_ws, size_t ws_size,
                              hipStream_t stream) {
    const float* x    = (const float*)d_in[0];
    const float* W_ih = (const float*)d_in[1];
    const float* W_hh = (const float*)d_in[2];
    const float* b_ih = (const float*)d_in[3];
    const float* b_hh = (const float*)d_in[4];
    const float* W_fc = (const float*)d_in[5];
    const float* b_fc = (const float*)d_in[6];
    float* out = (float*)d_out;

    // Workspace carve-up (~30 MB total)
    char* ws = (char*)d_ws;
    size_t off = 0;
    auto carve = [&](size_t bytes) -> void* {
        void* p = ws + off;
        off += (bytes + 255) & ~(size_t)255;
        return p;
    };
    unsigned short* xbf  = (unsigned short*)carve((size_t)B_ * T_ * I_ * 2);
    unsigned short* Wihb = (unsigned short*)carve((size_t)H_ * I_ * 2);
    unsigned short* Whhb = (unsigned short*)carve((size_t)H_ * H_ * 2);
    float*          bias = (float*)carve((size_t)H_ * 4);
    unsigned short* hb0  = (unsigned short*)carve((size_t)B_ * H_ * 2);
    unsigned short* hb1  = (unsigned short*)carve((size_t)B_ * H_ * 2);

    // Prologue: precision conversion + init
    {
        int n = B_ * T_ * I_;
        f32_to_bf16_kernel<<<(n + 255) / 256, 256, 0, stream>>>(x, xbf, n);
    }
    {
        int n = H_ * I_;
        f32_to_bf16_kernel<<<(n + 255) / 256, 256, 0, stream>>>(W_ih, Wihb, n);
    }
    {
        int n = H_ * H_;
        f32_to_bf16_kernel<<<(n + 255) / 256, 256, 0, stream>>>(W_hh, Whhb, n);
    }
    bias_combine_kernel<<<(H_ + 255) / 256, 256, 0, stream>>>(b_ih, b_hh, bias, H_);
    {
        int n = (B_ * H_) / 2; // zero hb0 as u32
        zero32_kernel<<<(n + 255) / 256, 256, 0, stream>>>((unsigned int*)hb0, n);
    }

    // Serial recurrence: 224 graph-captured step kernels, ping-pong h buffers
    unsigned short* hin  = hb0;
    unsigned short* hout = hb1;
    for (int t = 0; t < T_; ++t) {
        rnn_step_kernel<<<32, 256, 0, stream>>>(xbf, Wihb, Whhb, bias, hin, hout, t);
        unsigned short* tmp = hin; hin = hout; hout = tmp;
    }

    // Final FC on last hidden state (now in `hin` after the swap)
    fc_kernel<<<(B_ * O_ + 127) / 128, 128, 0, stream>>>(hin, W_fc, b_fc, out);
}
